// GNNInteractionNetwork_9612136808837
// MI455X (gfx1250) — matrix operand order repaired and checked by reading, not compiled, not run on hardware
//
#include <hip/hip_runtime.h>
#include <math.h>

typedef __attribute__((ext_vector_type(16))) _Float16 v16h;
typedef __attribute__((ext_vector_type(8)))  float    v8f;

#define HIDDEN 64

// One wave32 per (b, i). Loop over j in tiles of 16 edges.
// GEMM per tile: H2[16x64] = H1[16x64] x W2^T[64x64] via 8x v_wmma_f32_16x16x32_f16,
// b2 folded into the accumulator init, h1 built with packed-f16 VALU,
// relu via v_med3, row-broadcasts via raw ds_bpermute with hoisted indices.
__global__ __launch_bounds__(256)
void gnn_edge_wmma_kernel(const float* __restrict__ pos,
                          const float* __restrict__ eps_p,
                          const float* __restrict__ W1,
                          const float* __restrict__ b1,
                          const float* __restrict__ W2,
                          const float* __restrict__ b2,
                          const float* __restrict__ W3,
                          const float* __restrict__ b3_p,
                          float* __restrict__ out,
                          int B, int N)
{
    const int lane  = (int)(threadIdx.x & 31u);
    const int gwave = (int)((blockIdx.x * blockDim.x + threadIdx.x) >> 5);
    if (gwave >= B * N) return;           // wave-uniform exit (EXEC stays all-1s)
    const int b = gwave / N;
    const int i = gwave - b * N;

    const int row = lane & 15;            // tile row (j offset) / N-lane index
    const int hi  = lane >> 4;            // half-wave id

    const float eps   = eps_p[0];
    const float eps2  = eps * eps;
    const float b3    = b3_p[0];
    const float b3row = (row == 0) ? b3 : 0.0f;   // inject b3 once per row
    const float RELU_HI = 3.0e38f;

    const float* posB = pos + (size_t)b * N * 3;
    const float pix = posB[i*3+0], piy = posB[i*3+1], piz = posB[i*3+2];

    // ---- W1/b1 pre-converted to f16 in A-fragment order (ISA 16-bit A 16x32 layout) ----
    // element e: v=e>>1, k = kc*32 + (v&4 ? 16:0) + hi*8 + (v&3)*2 + (e&1)
    v16h w1h[2], b1h[2];
#pragma unroll
    for (int kc = 0; kc < 2; ++kc)
#pragma unroll
      for (int e = 0; e < 16; ++e) {
        const int v = e >> 1;
        const int k = kc*32 + ((v & 4) ? 16 : 0) + hi*8 + (v & 3)*2 + (e & 1);
        w1h[kc][e] = (_Float16)W1[k];
        b1h[kc][e] = (_Float16)b1[k];
      }

    // ---- B fragments (loop-invariant): B[k][n] = W2[n][k] ----
    // dense f16 B 32x16: elem e of half hi <-> k = (e&8 ? 16:0) + hi*8 + (e&7)
    v16h bfrag[2][4];
#pragma unroll
    for (int kc = 0; kc < 2; ++kc)
#pragma unroll
      for (int nc = 0; nc < 4; ++nc) {
        const int n = nc*16 + row;
#pragma unroll
        for (int e = 0; e < 16; ++e) {
          const int k = kc*32 + ((e & 8) ? 16 : 0) + hi*8 + (e & 7);
          bfrag[kc][nc][e] = (_Float16)W2[n*HIDDEN + k];
        }
      }

    // ---- b2 baked into accumulator init (C/D layout: N = lane, same for all 8 VGPRs) ----
    float w3f[4];
    v8f   cinit[4];
#pragma unroll
    for (int nc = 0; nc < 4; ++nc) {
      const float b2v = b2[nc*16 + row];
      w3f[nc] = W3[nc*16 + row];
#pragma unroll
      for (int r = 0; r < 8; ++r) cinit[nc][r] = b2v;
    }

    // loop-invariant bpermute byte-indices: lane holding row hi*8+r in this half
    int srcb[8];
#pragma unroll
    for (int r = 0; r < 8; ++r) srcb[r] = (((lane & 16) + hi*8 + r) << 2);

    const v16h zeroh = {};
    float ax = 0.f, ay = 0.f, az = 0.f;

    for (int j0 = 0; j0 < N; j0 += 16) {
      const int j = j0 + row;
      const float dx = posB[j*3+0] - pix;
      const float dy = posB[j*3+1] - piy;
      const float dz = posB[j*3+2] - piz;
      const float r2 = fmaf(dx,dx, fmaf(dy,dy, dz*dz)) + ((j == i) ? 1.0f : 0.0f);

      // h1 = relu(r2*W1 + b1) built in packed f16 directly in A-fragment layout
      const _Float16 r2h = (_Float16)r2;
      v16h r2v;
#pragma unroll
      for (int e = 0; e < 16; ++e) r2v[e] = r2h;
      v16h a0 = __builtin_elementwise_max(
                    __builtin_elementwise_fma(r2v, w1h[0], b1h[0]), zeroh);
      v16h a1 = __builtin_elementwise_max(
                    __builtin_elementwise_fma(r2v, w1h[1], b1h[1]), zeroh);

      // 4 independent accumulation chains; C starts at b2
      v8f c[4];
#pragma unroll
      for (int nc = 0; nc < 4; ++nc)
        c[nc] = __builtin_amdgcn_wmma_f32_16x16x32_f16(false, a0, false, bfrag[0][nc],
                                                       (short)0, cinit[nc], false, false);
#pragma unroll
      for (int nc = 0; nc < 4; ++nc)
        c[nc] = __builtin_amdgcn_wmma_f32_16x16x32_f16(false, a1, false, bfrag[1][nc],
                                                       (short)0, c[nc], false, false);

      // gated phys * dx for this lane's own j-row; phys = rsq(p)^3
      const float p    = r2 + eps2;
      const float rq   = __builtin_amdgcn_rsqf(p);
      const float phys = rq * rq * rq;
      const float g    = (j == i) ? 0.0f : phys;
      const float cx = g * dx, cy = g * dy, cz = g * dz;

      // issue all row-broadcasts up front (single dscnt drain, overlaps epilogue VALU)
      int rxI[8], ryI[8], rzI[8];
      const int cxI = __float_as_int(cx);
      const int cyI = __float_as_int(cy);
      const int czI = __float_as_int(cz);
#pragma unroll
      for (int r = 0; r < 8; ++r) {
        rxI[r] = __builtin_amdgcn_ds_bpermute(srcb[r], cxI);
        ryI[r] = __builtin_amdgcn_ds_bpermute(srcb[r], cyI);
        rzI[r] = __builtin_amdgcn_ds_bpermute(srcb[r], czI);
      }

      // per-lane partial of m[row]: rows hi*8+r, n = nc*16+(lane&15)
      // relu as single v_med3 (avoids canonicalize+max pair on WMMA output)
      float s[8];
#pragma unroll
      for (int r = 0; r < 8; ++r) s[r] = b3row;
#pragma unroll
      for (int nc = 0; nc < 4; ++nc)
#pragma unroll
        for (int r = 0; r < 8; ++r)
          s[r] = fmaf(w3f[nc], __builtin_amdgcn_fmed3f(c[nc][r], 0.0f, RELU_HI), s[r]);

      // acc += s[r] * (phys*dx)[row = hi*8 + r]   (force is linear in m partials)
#pragma unroll
      for (int r = 0; r < 8; ++r) {
        ax = fmaf(s[r], __int_as_float(rxI[r]), ax);
        ay = fmaf(s[r], __int_as_float(ryI[r]), ay);
        az = fmaf(s[r], __int_as_float(rzI[r]), az);
      }
    }

    // full-wave reduction of partial accelerations
#pragma unroll
    for (int m = 16; m >= 1; m >>= 1) {
      ax += __shfl_xor(ax, m, 32);
      ay += __shfl_xor(ay, m, 32);
      az += __shfl_xor(az, m, 32);
    }
    if (lane == 0) {
      float* o = out + ((size_t)b * N + i) * 3;
      o[0] = ax; o[1] = ay; o[2] = az;
    }
}

extern "C" void kernel_launch(void* const* d_in, const int* in_sizes, int n_in,
                              void* d_out, int out_size, void* d_ws, size_t ws_size,
                              hipStream_t stream)
{
    const float* pos = (const float*)d_in[0];
    const float* eps = (const float*)d_in[1];
    const float* W1  = (const float*)d_in[2];
    const float* b1  = (const float*)d_in[3];
    const float* W2  = (const float*)d_in[4];
    const float* b2  = (const float*)d_in[5];
    const float* W3  = (const float*)d_in[6];
    const float* b3  = (const float*)d_in[7];
    float* out = (float*)d_out;

    const int B = 4;
    const int N = in_sizes[0] / (B * 3);   // pos is [B, N, 3]
    const int totalWaves = B * N;          // one wave per (b, i)
    const int threads = 256;               // 8 wave32 per block
    const int blocks  = (totalWaves * 32 + threads - 1) / threads;

    gnn_edge_wmma_kernel<<<blocks, threads, 0, stream>>>(
        pos, eps, W1, b1, W2, b2, W3, b3, out, B, N);
}
